// SpiralNet_67422396612957
// MI455X (gfx1250) — matrix operand order repaired and checked by reading, not compiled, not run on hardware
//
#include <hip/hip_runtime.h>
#include <cstdint>
#include <cstddef>

// ---------------------------------------------------------------------------
// SpiralNet++ inference for MI455X (gfx1250, wave32, WMMA).
//
// All heavy GEMMs run through v_wmma_f32_16x16x32_f16 (f16 A/B, f32 acc).
// Weights are pre-packed into the ISA's per-lane B-fragment layout so each
// k-step is: 2x16B gathered A loads + 1x32B packed B load + 1 WMMA per n-tile.
// Gather addressing uses 32-bit v8h-element indices so the compiler emits
// saddr + scaled-voffset loads instead of 64-bit vector address arithmetic.
// ---------------------------------------------------------------------------

typedef __attribute__((ext_vector_type(16))) _Float16 v16h;
typedef __attribute__((ext_vector_type(8)))  _Float16 v8h;
typedef __attribute__((ext_vector_type(8)))  float    v8f;

__device__ __forceinline__ float elu1(float v) {
    return v > 0.0f ? v : (__expf(v) - 1.0f);
}

// ---------------------------------------------------------------------------
// Pack f32 weight matrix W[K, Cs] (row-major, in x out) into f16 WMMA
// B-fragments, padded to Cd output columns (Cd multiple of 16, K mult of 32).
// Packed element p = ((nt*KT + kt)*32 + lane)*16 + j  holds
//   B[k = kt*32 + (j>=8 ? 16:0) + (lane>=16 ? 8:0) + (j&7)][n = nt*16 + (lane&15)]
// matching the 16-bit 32x16 B layout (lanes 0-15: K=0..7,16..23, lanes 16-31: 8..15,24..31).
// ---------------------------------------------------------------------------
__global__ void pack_weights_kernel(const float* __restrict__ W,
                                    _Float16* __restrict__ Wp,
                                    int K, int Cs, int Cd) {
    int p = blockIdx.x * blockDim.x + threadIdx.x;
    int total = K * Cd;
    if (p >= total) return;
    int j    = p & 15;
    int lane = (p >> 4) & 31;
    int rest = p >> 9;
    int KT   = K >> 5;
    int kt   = rest % KT;
    int nt   = rest / KT;
    int k = kt * 32 + ((j >> 3) << 4) + ((lane >> 4) << 3) + (j & 7);
    int n = nt * 16 + (lane & 15);
    float v = (n < Cs) ? W[(size_t)k * Cs + n] : 0.0f;
    Wp[p] = (_Float16)v;
}

// ---------------------------------------------------------------------------
// Layer 0: x1 = f16( elu(pos @ W0 + b0) ),  K=3 -> scalar is optimal.
// ---------------------------------------------------------------------------
__global__ void layer0_kernel(const float* __restrict__ pos,
                              const float* __restrict__ W0,
                              const float* __restrict__ b0,
                              _Float16* __restrict__ x1, int Nv) {
    int i = blockIdx.x * blockDim.x + threadIdx.x;
    if (i >= Nv) return;
    float px = pos[3 * i + 0];
    float py = pos[3 * i + 1];
    float pz = pos[3 * i + 2];
#pragma unroll
    for (int c = 0; c < 16; ++c) {
        float v = fmaf(px, W0[c], fmaf(py, W0[16 + c], fmaf(pz, W0[32 + c], b0[c])));
        x1[(size_t)i * 16 + c] = (_Float16)elu1(v);
    }
}

// ---------------------------------------------------------------------------
// Spiral-gather / dense GEMM via WMMA.
//   Y[m, n] = act( sum_k A[m,k] * W[k,n] + b[n] )
//   A[m,k] = X[ idx[m, k/CIN] ][ k%CIN ]   (GATHER)  or  X[m,k] (dense)
// One wave -> 16 rows x (16*NTN) cols. 256 threads = 8 waves per block.
// ---------------------------------------------------------------------------
template <int CIN, int COUT, int SLEN, int NTN, bool GATHER, bool DO_ELU, bool OUT_F16>
__global__ void spiral_gemm_kernel(const _Float16* __restrict__ X,
                                   const int* __restrict__ idx,
                                   const _Float16* __restrict__ Wp,
                                   const float* __restrict__ bias, int nbias,
                                   void* __restrict__ Out, int Nv) {
    constexpr int K    = CIN * SLEN;        // reduction length
    constexpr int KT   = K / 32;            // k-tiles of 32
    constexpr int NTW  = COUT / (16 * NTN); // n-tile groups per row-tile
    constexpr int CV8  = CIN / 8;           // row length in v8h (16B) units

    const int wave = threadIdx.x >> 5;
    const int lane = threadIdx.x & 31;
    const int tile = blockIdx.x * 8 + wave;
    const int mt   = tile / NTW;
    const int ntg  = tile - mt * NTW;
    if (mt >= (Nv >> 4)) return;            // wave-uniform guard (EXEC stays full)

    const int m0     = mt << 4;
    const int m      = m0 + (lane & 15);
    const int kb_off = (lane >> 4) << 3;    // lanes 16-31 start at K+8
    const int ncol   = lane & 15;

    v8f acc[NTN] = {};

    // X viewed as 16B elements -> 32-bit scaled-offset addressing (GVS mode).
    const v8h* __restrict__ Xv = (const v8h*)X;

    // Base of this wave's packed-B fragments; advance 512 halves per k-tile.
    const _Float16* wptr = Wp + (size_t)(ntg * NTN) * KT * 512 + (size_t)lane * 16;

    for (int kt = 0; kt < KT; ++kt) {
        const int kb = (kt << 5) + kb_off;
        v8h lo, hi;
        if constexpr (GATHER) {
            // Each 8-half run lies inside one spiral element (CIN mult of 16,
            // run start mult of 8) -> one aligned 16B gathered load each.
            const int s1 = kb / CIN,        c1 = kb & (CIN - 1);
            const int s2 = (kb + 16) / CIN, c2 = (kb + 16) & (CIN - 1);
            const uint32_t r1 = (uint32_t)idx[m * SLEN + s1];
            const uint32_t r2 = (uint32_t)idx[m * SLEN + s2];
            lo = Xv[r1 * CV8 + (uint32_t)(c1 >> 3)];
            hi = Xv[r2 * CV8 + (uint32_t)(c2 >> 3)];
        } else {
            const uint32_t base = (uint32_t)m * CV8;
            lo = Xv[base + (uint32_t)(kb >> 3)];
            hi = Xv[base + (uint32_t)((kb + 16) >> 3)];
        }
        // 16-bit A 16x32 layout: halves 0-7 = K kb..kb+7, halves 8-15 = kb+16..kb+23
        v16h a = __builtin_shufflevector(lo, hi, 0, 1, 2, 3, 4, 5, 6, 7,
                                                 8, 9, 10, 11, 12, 13, 14, 15);
#pragma unroll
        for (int t = 0; t < NTN; ++t) {
            v16h b = *(const v16h*)(wptr + (size_t)t * KT * 512 + (size_t)kt * 512);
            acc[t] = __builtin_amdgcn_wmma_f32_16x16x32_f16(
                false, a, false, b, (short)0, acc[t], false, false);
        }
    }

    // Epilogue. D layout: VGPR j -> row j (lanes 0-15) / j+8 (lanes 16-31).
    const int row_half = (lane >> 4) << 3;
#pragma unroll
    for (int t = 0; t < NTN; ++t) {
        const int n  = (ntg * NTN + t) * 16 + ncol;
        const float bn = (n < nbias) ? bias[n] : 0.0f;
#pragma unroll
        for (int j = 0; j < 8; ++j) {
            const int row = m0 + j + row_half;
            float v = acc[t][j] + bn;
            if constexpr (DO_ELU) v = elu1(v);
            if constexpr (OUT_F16)
                ((_Float16*)Out)[(size_t)row * COUT + n] = (_Float16)v;
            else
                ((float*)Out)[(size_t)row * COUT + n] = v;
        }
    }
}

// ---------------------------------------------------------------------------
// log_softmax over the 10 valid columns of the padded [N,16] logits buffer.
// ---------------------------------------------------------------------------
__global__ void logsoftmax_kernel(const float* __restrict__ logits,
                                  float* __restrict__ out, int Nv) {
    int i = blockIdx.x * blockDim.x + threadIdx.x;
    if (i >= Nv) return;
    float v[10];
    float mx = -3.0e38f;
#pragma unroll
    for (int c = 0; c < 10; ++c) {
        v[c] = logits[(size_t)i * 16 + c];
        mx = fmaxf(mx, v[c]);
    }
    float s = 0.0f;
#pragma unroll
    for (int c = 0; c < 10; ++c) s += __expf(v[c] - mx);
    float l = __logf(s);
#pragma unroll
    for (int c = 0; c < 10; ++c) out[(size_t)i * 10 + c] = v[c] - mx - l;
}

// ---------------------------------------------------------------------------
extern "C" void kernel_launch(void* const* d_in, const int* in_sizes, int n_in,
                              void* d_out, int out_size, void* d_ws, size_t ws_size,
                              hipStream_t stream) {
    const float* pos = (const float*)d_in[0];
    const int*   idx = (const int*)  d_in[1];
    const float* W0  = (const float*)d_in[2];
    const float* b0  = (const float*)d_in[3];
    const float* W1  = (const float*)d_in[4];
    const float* b1  = (const float*)d_in[5];
    const float* W2  = (const float*)d_in[6];
    const float* b2  = (const float*)d_in[7];
    const float* W3  = (const float*)d_in[8];
    const float* b3  = (const float*)d_in[9];
    const float* Wf1 = (const float*)d_in[10];
    const float* bf1 = (const float*)d_in[11];
    const float* Wf2 = (const float*)d_in[12];
    const float* bf2 = (const float*)d_in[13];
    float* out = (float*)d_out;

    const int Nv = in_sizes[0] / 3;   // 100000 (multiple of 16)

    // ---- workspace layout (256B aligned) ----
    char* ws = (char*)d_ws;
    size_t off = 0;
    auto alloc = [&](size_t bytes) -> void* {
        void* p = ws + off;
        off = (off + bytes + 255) & ~(size_t)255;
        return p;
    };
    _Float16* x1     = (_Float16*)alloc((size_t)Nv * 16 * 2);
    _Float16* x2     = (_Float16*)alloc((size_t)Nv * 32 * 2);
    _Float16* x3     = (_Float16*)alloc((size_t)Nv * 64 * 2);
    _Float16* x4     = (_Float16*)alloc((size_t)Nv * 128 * 2);
    _Float16* x5     = (_Float16*)alloc((size_t)Nv * 256 * 2);
    float*    logits = (float*)   alloc((size_t)Nv * 16 * 4);
    _Float16* W1p    = (_Float16*)alloc((size_t)320  * 32  * 2);
    _Float16* W2p    = (_Float16*)alloc((size_t)640  * 64  * 2);
    _Float16* W3p    = (_Float16*)alloc((size_t)1280 * 128 * 2);
    _Float16* Wf1p   = (_Float16*)alloc((size_t)128  * 256 * 2);
    _Float16* Wf2p   = (_Float16*)alloc((size_t)256  * 16  * 2);
    (void)ws_size;

    // ---- pack weights into WMMA B-fragment layout ----
    pack_weights_kernel<<<(320 * 32 + 255) / 256, 256, 0, stream>>>(W1, W1p, 320, 32, 32);
    pack_weights_kernel<<<(640 * 64 + 255) / 256, 256, 0, stream>>>(W2, W2p, 640, 64, 64);
    pack_weights_kernel<<<(1280 * 128 + 255) / 256, 256, 0, stream>>>(W3, W3p, 1280, 128, 128);
    pack_weights_kernel<<<(128 * 256 + 255) / 256, 256, 0, stream>>>(Wf1, Wf1p, 128, 256, 256);
    pack_weights_kernel<<<(256 * 16 + 255) / 256, 256, 0, stream>>>(Wf2, Wf2p, 256, 10, 16);

    // ---- layer 0 (K=3, scalar) ----
    layer0_kernel<<<(Nv + 255) / 256, 256, 0, stream>>>(pos, W0, b0, x1, Nv);

    const int mtiles = Nv / 16; // 6250

    // ---- spiral conv layers (WMMA) ----
    {   // L1: [N,320] @ [320,32]
        int tiles = mtiles * (32 / 32);
        spiral_gemm_kernel<16, 32, 20, 2, true, true, true>
            <<<(tiles + 7) / 8, 256, 0, stream>>>(x1, idx, W1p, b1, 32, x2, Nv);
    }
    {   // L2: [N,640] @ [640,64]
        int tiles = mtiles * (64 / 32);
        spiral_gemm_kernel<32, 64, 20, 2, true, true, true>
            <<<(tiles + 7) / 8, 256, 0, stream>>>(x2, idx, W2p, b2, 64, x3, Nv);
    }
    {   // L3: [N,1280] @ [1280,128]
        int tiles = mtiles * (128 / 32);
        spiral_gemm_kernel<64, 128, 20, 2, true, true, true>
            <<<(tiles + 7) / 8, 256, 0, stream>>>(x3, idx, W3p, b3, 128, x4, Nv);
    }
    {   // FC1: [N,128] @ [128,256]
        int tiles = mtiles * (256 / 32);
        spiral_gemm_kernel<128, 256, 1, 2, false, true, true>
            <<<(tiles + 7) / 8, 256, 0, stream>>>(x4, nullptr, Wf1p, bf1, 256, x5, Nv);
    }
    {   // FC2: [N,256] @ [256,16(pad)] -> f32 logits, no activation
        int tiles = mtiles * 1;
        spiral_gemm_kernel<256, 16, 1, 1, false, false, false>
            <<<(tiles + 7) / 8, 256, 0, stream>>>(x5, nullptr, Wf2p, bf2, 10, logits, Nv);
    }

    // ---- log_softmax -> d_out [N,10] f32 ----
    logsoftmax_kernel<<<(Nv + 255) / 256, 256, 0, stream>>>(logits, out, Nv);
}
